// Model_86852828660360
// MI455X (gfx1250) — compile-verified, matrix operand order checked by reading, not run on hardware
//
#include <hip/hip_runtime.h>
#include <stdint.h>

// ---------------- problem constants ----------------
#define BATCH   128
#define DMODEL  512
#define G3      (3 * DMODEL)     // 1536 gate columns (r | z | n)
#define TSEQ    512
#define ENCIN   21
#define PREDLEN 96

// persistent-kernel shape: 768 wave32 tiles over the [128 x 1536] gate matrix
#define NBLK     96
#define NTHREADS 256
#define NLANES   (NBLK * NTHREADS)   // 24576

// ---------------- WMMA vector types (gfx1250, wave32) ----------------
typedef __bf16          v16bf __attribute__((ext_vector_type(16)));
typedef float           v8f   __attribute__((ext_vector_type(8)));
typedef unsigned short  v8us  __attribute__((ext_vector_type(8)));
typedef unsigned short  v16us __attribute__((ext_vector_type(16)));

union BF16x16 { v16us u; v16bf b; };

__device__ __forceinline__ unsigned short f2bf(float f) {
    unsigned int u = __float_as_uint(f);
    u += 0x7fffu + ((u >> 16) & 1u);          // round-to-nearest-even
    return (unsigned short)(u >> 16);
}
__device__ __forceinline__ float bf2f(unsigned short h) {
    return __uint_as_float(((unsigned int)h) << 16);
}

// Load one 16x32 bf16 WMMA fragment. For 16-bit A (ISA 7.12.2): lane%16 = M,
// lane half selects K-subgroup; each lane needs two contiguous 8-element
// (16-byte) runs -> two global_load_b128. With weights stored [N][K] row-major
// the transposed-B fragment uses the identical addressing with rows = N.
__device__ __forceinline__ v16bf load_frag(const unsigned short* __restrict__ base,
                                           size_t row_stride, int row, int kbase, int lane) {
    const unsigned short* p = base + (size_t)row * row_stride
                                   + (size_t)(kbase + ((lane >> 4) << 3));
    v8us lo = *(const v8us*)p;
    v8us hi = *(const v8us*)(p + 16);
    BF16x16 f;
    f.u = __builtin_shufflevector(lo, hi, 0,1,2,3,4,5,6,7,8,9,10,11,12,13,14,15);
    return f.b;
}

__device__ __forceinline__ float sigmoidf_(float x) { return 1.0f / (1.0f + __expf(-x)); }

// Device-wide split barrier: monotonic counter, agent-scope release/acquire
// atomics (compiler emits the gfx1250 cache invalidate/writeback ops), s_sleep
// while spinning. All 96 blocks are grid-resident by construction.
__device__ __forceinline__ void grid_sync(int* bar, int& phase) {
    ++phase;
    __threadfence();                 // make this thread's stores agent-visible
    __syncthreads();
    if (threadIdx.x == 0) {
        __hip_atomic_fetch_add(bar, 1, __ATOMIC_RELEASE, __HIP_MEMORY_SCOPE_AGENT);
        while (__hip_atomic_load(bar, __ATOMIC_ACQUIRE, __HIP_MEMORY_SCOPE_AGENT)
               < phase * NBLK)
            __builtin_amdgcn_s_sleep(1);
    }
    __syncthreads();
    __threadfence();                 // invalidate near caches before consuming
}

// ---------------- persistent GRU layer sweep ----------------
// Phase A (per step t): every wave owns one 16x16 tile of the [128 x 1536]
//   gate matrix. gh = h_prev @ W_hh^T via 16 x v_wmma_f32_16x16x32_bf16
//   (K = 512). Layer1 also computes gx = h0[:,t,:] @ W_ih1^T with a second
//   WMMA accumulator; layer0's gx is a K=21 VALU dot against x_enc.
//   r,z chunks store gx+gh (+biases); n chunk stores gx / gh separately
//   (PyTorch GRU: n = tanh(xn + b_in + r*(hn + b_hn))).
// Phase B: elementwise gate math + in-place bf16 h update; layer0 also
//   records h into h0_seq (bf16), layer1 records the last 96 steps in f32.
template<int LAYER>
__global__ void __launch_bounds__(NTHREADS, 1)
gru_persistent(const float* __restrict__ x_enc,
               const float* __restrict__ W_ih0,     // [1536][21] f32 (layer0)
               const float* __restrict__ b_ih,
               const float* __restrict__ b_hh,
               const unsigned short* __restrict__ Wih_bf,  // [1536][512] bf16 (layer1)
               const unsigned short* __restrict__ Whh_bf,  // [1536][512] bf16
               const unsigned short* __restrict__ h0_seq,  // [128][512][512] bf16 (layer1 in)
               unsigned short* __restrict__ h0_seq_out,    // (layer0 out, same buffer)
               unsigned short* __restrict__ h,             // [128][512] bf16, in-place
               float* __restrict__ sumbuf,                 // [128][1024] r|z pre-activations
               float* __restrict__ gxn,                    // [128][512]
               float* __restrict__ ghn,                    // [128][512]
               float* __restrict__ h1_last,                // [128][96][512] f32 (layer1)
               int* __restrict__ bar)
{
    const int lane = threadIdx.x & 31;
    const int wave = threadIdx.x >> 5;
    const int wgid = blockIdx.x * (NTHREADS / 32) + wave;  // 0..767
    const int mt   = wgid & 7;                             // batch tile 0..7
    const int nt   = wgid >> 3;                            // gate-col tile 0..95
    const int col  = nt * 16 + (lane & 15);                // 0..1535 (uniform gate chunk per wave)
    const int r0   = mt * 16 + ((lane >> 4) << 3);         // C-fragment row base (ISA 7.12.2)
    const int arow = mt * 16 + (lane & 15);                // A-fragment row (M = lane%16)
    const int gid  = blockIdx.x * NTHREADS + threadIdx.x;
    const float bih = b_ih[col];
    const float bhh = b_hh[col];
    int phase = 0;

    for (int t = 0; t < TSEQ; ++t) {
        // ---------------- Phase A: gate GEMMs ----------------
        v8f gha = {};
        v8f gxa = {};
#pragma unroll 4
        for (int kk = 0; kk < DMODEL; kk += 32) {
            v16bf a = load_frag(h, DMODEL, arow, kk, lane);
            v16bf b = load_frag(Whh_bf, DMODEL, col, kk, lane);
            gha = __builtin_amdgcn_wmma_f32_16x16x32_bf16(
                      false, a, false, b, (short)0, gha, false, false);
            if (LAYER == 1) {
                v16bf a0 = load_frag(h0_seq + (size_t)t * DMODEL,
                                     (size_t)TSEQ * DMODEL, arow, kk, lane);
                v16bf b0 = load_frag(Wih_bf, DMODEL, col, kk, lane);
                gxa = __builtin_amdgcn_wmma_f32_16x16x32_bf16(
                          false, a0, false, b0, (short)0, gxa, false, false);
            }
        }

#pragma unroll
        for (int i = 0; i < 8; ++i) {
            const int row = r0 + i;                       // batch index 0..127
            const float ghv = gha[i] + bhh;
            float gxv;
            if (LAYER == 1) {
                gxv = gxa[i] + bih;
            } else {                                      // K=21 input dot, L2-resident
                const float* xp = x_enc + ((size_t)row * TSEQ + t) * ENCIN;
                const float* wp = W_ih0 + (size_t)col * ENCIN;
                float s = bih;
#pragma unroll
                for (int k = 0; k < ENCIN; ++k) s += xp[k] * wp[k];
                gxv = s;
            }
            if (col < 2 * DMODEL) {                       // r and z: only sum needed
                sumbuf[(size_t)row * (2 * DMODEL) + col] = gxv + ghv;
            } else {                                      // n: keep separate for r*(...)
                const int j = col - 2 * DMODEL;
                gxn[(size_t)row * DMODEL + j] = gxv;
                ghn[(size_t)row * DMODEL + j] = ghv;
            }
        }
        grid_sync(bar, phase);

        // ---------------- Phase B: elementwise h update ----------------
        for (int idx = gid; idx < BATCH * DMODEL; idx += NLANES) {
            const int bb = idx >> 9;                      // /512
            const int j  = idx & (DMODEL - 1);
            const float r = sigmoidf_(sumbuf[(size_t)bb * (2 * DMODEL) + j]);
            const float z = sigmoidf_(sumbuf[(size_t)bb * (2 * DMODEL) + DMODEL + j]);
            const float n = tanhf(gxn[idx] + r * ghn[idx]);
            const float hp = bf2f(h[idx]);
            const float hn = (1.0f - z) * n + z * hp;
            h[idx] = f2bf(hn);
            if (LAYER == 0) {
                h0_seq_out[((size_t)bb * TSEQ + t) * DMODEL + j] = f2bf(hn);
            } else if (t >= TSEQ - PREDLEN) {
                h1_last[((size_t)bb * PREDLEN + (t - (TSEQ - PREDLEN))) * DMODEL + j] = hn;
            }
        }
        grid_sync(bar, phase);
    }
}

// ---------------- helpers ----------------
__global__ void convert_bf16_kernel(const float* __restrict__ src,
                                    unsigned short* __restrict__ dst, int n) {
    int i = blockIdx.x * blockDim.x + threadIdx.x;
    const int stride = gridDim.x * blockDim.x;
    for (; i < n; i += stride) dst[i] = f2bf(src[i]);
}

__global__ void init_state_kernel(unsigned int* __restrict__ hwords, int nwords,
                                  int* __restrict__ bar) {
    int i = blockIdx.x * blockDim.x + threadIdx.x;
    if (i == 0) *bar = 0;
    const int stride = gridDim.x * blockDim.x;
    for (; i < nwords; i += stride) hwords[i] = 0u;
}

// Final 512 -> 21 projection of the last 96 steps. One block per (b, t) row:
// stage the h row in LDS once, 21 lanes each run a K=512 dot from LDS.
__global__ void proj_kernel(const float* __restrict__ h1,
                            const float* __restrict__ Wp,
                            const float* __restrict__ bp,
                            float* __restrict__ out) {
    __shared__ float sh[DMODEL];
    const int row = blockIdx.x;                           // 0 .. 128*96-1
    const float* hr = h1 + (size_t)row * DMODEL;
    for (int k = threadIdx.x; k < DMODEL; k += blockDim.x) sh[k] = hr[k];
    __syncthreads();
    const int o = threadIdx.x;
    if (o < ENCIN) {
        const float* w = Wp + (size_t)o * DMODEL;
        float s = bp[o];
#pragma unroll 8
        for (int k = 0; k < DMODEL; ++k) s += sh[k] * w[k];
        out[(size_t)row * ENCIN + o] = s;
    }
}

// ---------------- launcher ----------------
extern "C" void kernel_launch(void* const* d_in, const int* in_sizes, int n_in,
                              void* d_out, int out_size, void* d_ws, size_t ws_size,
                              hipStream_t stream) {
    (void)in_sizes; (void)n_in; (void)out_size; (void)ws_size;

    const float* x_enc  = (const float*)d_in[0];
    const float* W_ih0  = (const float*)d_in[4];
    const float* W_hh0  = (const float*)d_in[5];
    const float* b_ih0  = (const float*)d_in[6];
    const float* b_hh0  = (const float*)d_in[7];
    const float* W_ih1  = (const float*)d_in[8];
    const float* W_hh1  = (const float*)d_in[9];
    const float* b_ih1  = (const float*)d_in[10];
    const float* b_hh1  = (const float*)d_in[11];
    const float* W_proj = (const float*)d_in[12];
    const float* b_proj = (const float*)d_in[13];
    float* out = (float*)d_out;

    // workspace carve-out (~97 MB total), 256B aligned slices
    char* ws = (char*)d_ws;
    size_t off = 0;
    auto alloc = [&](size_t bytes) -> void* {
        void* p = ws + off;
        off += (bytes + 255) & ~(size_t)255;
        return p;
    };
    int*            bar      = (int*)alloc(256);
    unsigned short* Whh0_bf  = (unsigned short*)alloc((size_t)G3 * DMODEL * 2);
    unsigned short* Wih1_bf  = (unsigned short*)alloc((size_t)G3 * DMODEL * 2);
    unsigned short* Whh1_bf  = (unsigned short*)alloc((size_t)G3 * DMODEL * 2);
    unsigned short* hbuf     = (unsigned short*)alloc((size_t)BATCH * DMODEL * 2);
    float*          sumbuf   = (float*)alloc((size_t)BATCH * 2 * DMODEL * 4);
    float*          gxn      = (float*)alloc((size_t)BATCH * DMODEL * 4);
    float*          ghn      = (float*)alloc((size_t)BATCH * DMODEL * 4);
    unsigned short* h0_seq   = (unsigned short*)alloc((size_t)BATCH * TSEQ * DMODEL * 2);
    float*          h1_last  = (float*)alloc((size_t)BATCH * PREDLEN * DMODEL * 4);

    // one-time (per launch) weight down-conversion: 2.4M elems, HBM-trivial
    convert_bf16_kernel<<<512, 256, 0, stream>>>(W_hh0, Whh0_bf, G3 * DMODEL);
    convert_bf16_kernel<<<512, 256, 0, stream>>>(W_ih1, Wih1_bf, G3 * DMODEL);
    convert_bf16_kernel<<<512, 256, 0, stream>>>(W_hh1, Whh1_bf, G3 * DMODEL);

    // layer 0 sweep (h := 0, barrier := 0, then persistent kernel over T)
    init_state_kernel<<<64, 256, 0, stream>>>((unsigned int*)hbuf, BATCH * DMODEL / 2, bar);
    gru_persistent<0><<<NBLK, NTHREADS, 0, stream>>>(
        x_enc, W_ih0, b_ih0, b_hh0, nullptr, Whh0_bf,
        nullptr, h0_seq, hbuf, sumbuf, gxn, ghn, nullptr, bar);

    // layer 1 sweep
    init_state_kernel<<<64, 256, 0, stream>>>((unsigned int*)hbuf, BATCH * DMODEL / 2, bar);
    gru_persistent<1><<<NBLK, NTHREADS, 0, stream>>>(
        nullptr, nullptr, b_ih1, b_hh1, Wih1_bf, Whh1_bf,
        h0_seq, nullptr, hbuf, sumbuf, gxn, ghn, h1_last, bar);

    // projection of the last 96 steps
    proj_kernel<<<BATCH * PREDLEN, 64, 0, stream>>>(h1_last, W_proj, b_proj, out);
}
